// transformer_11441792877290
// MI455X (gfx1250) — compile-verified, matrix operand order checked by reading, not run on hardware
//
#include <hip/hip_runtime.h>
#include <hip/hip_bf16.h>
#include <cstdint>

typedef unsigned short u16;
typedef __attribute__((ext_vector_type(16))) __bf16 bf16x16;
typedef __attribute__((ext_vector_type(8)))  float  floatx8;

#define TM 128
#define TN 128
#define TK 32
#define LP 40   // LDS row pitch in u16 units (80 bytes: multiple of 16B -> aligned b128)

union FragBF { uint4 u[2]; bf16x16 v; };

__device__ __forceinline__ u16 f2bf(float f) {
  unsigned u = __float_as_uint(f);
  unsigned r = u + 0x7fffu + ((u >> 16) & 1u);   // round-to-nearest-even
  return (u16)(r >> 16);
}

// ---------------------------------------------------------------- convert f32 -> bf16
__global__ void cvt_f32_bf16(const float* __restrict__ x, u16* __restrict__ y, long long n) {
  long long i = (long long)blockIdx.x * blockDim.x + threadIdx.x;
  const long long stride = (long long)gridDim.x * blockDim.x;
  for (; i < n; i += stride) y[i] = f2bf(x[i]);
}

// ---------------------------------------------------------------- batched bf16 WMMA GEMM
// C[z][M,N] (f32) = A[z][M,K] (bf16, row stride lda) * B[z] (bf16, element (k,n) at k*ldbk + n*ldbn)
// epi: 0 = none, 1 = +bias[n], 2 = relu(x + bias[n])
// Requirements: M % 128 == 0, K % 32 == 0 (true for all calls); N may be ragged.
// All per-batch buffers are < 2^31 elements -> 32-bit offsets + uniform base (GVS addressing).
__global__ __launch_bounds__(256)
void wmma_gemm_bf16(const u16* __restrict__ A, const u16* __restrict__ Bm,
                    float* __restrict__ C,
                    int M, int N, int K,
                    int lda, int ldbk, int ldbn, int ldc,
                    long long sA, long long sB, long long sC,
                    const float* __restrict__ bias, int epi)
{
  __shared__ __align__(16) u16 As[TM * LP];
  __shared__ __align__(16) u16 Bs[TN * LP];   // stored transposed: Bs[n][k]

  const int tid  = threadIdx.x;
  const int lane = tid & 31;
  const int wid  = tid >> 5;            // 8 wave32 waves
  const int m0 = blockIdx.y * TM;
  const int n0 = blockIdx.x * TN;

  A  += (long long)blockIdx.z * sA;     // uniform per block
  Bm += (long long)blockIdx.z * sB;
  C  += (long long)blockIdx.z * sC;

  const int waveM = (wid & 3) * 32;     // 4x2 wave grid: 32 rows x 64 cols per wave
  const int waveN = (wid >> 2) * 64;

  const floatx8 zero8 = {};
  floatx8 acc[2][4];
#pragma unroll
  for (int i = 0; i < 2; ++i)
#pragma unroll
    for (int j = 0; j < 4; ++j) acc[i][j] = zero8;

  const int arow = tid >> 1;            // 0..127 : tile row (A) / tile col n (B)
  const int asel = (tid & 1) * 16;      // 0 or 16 : which 16-element K chunk

  // Branchless ragged-N: clamp column so loads are always in-bounds; zero via one AND.
  const int nn = n0 + arow;
  const unsigned bmask = (nn < N) ? 0xffffffffu : 0u;
  const int nc = (nn < N) ? nn : (N - 1);

  // 32-bit loop-carried offsets from uniform bases (independent per-load addresses).
  unsigned aoff = (unsigned)((m0 + arow) * lda + asel);
  unsigned boff = (unsigned)(nc * ldbn + asel * ldbk);
  const unsigned bstep = (unsigned)(TK * ldbk);
  unsigned jo[16];                      // hoisted strided-column offsets: j*ldbk
#pragma unroll
  for (int j = 0; j < 16; ++j) jo[j] = (unsigned)(j * ldbk);

  for (int k0 = 0; k0 < K; k0 += TK) {
    // --- global loads (A: contiguous b128 x2; B: 16 independent strided u16, masked)
    const uint4 av0 = *(const uint4*)(A + aoff);
    const uint4 av1 = *(const uint4*)(A + aoff + 8);

    unsigned bw[8];
#pragma unroll
    for (int j = 0; j < 8; ++j) {
      const unsigned lo = (unsigned)Bm[boff + jo[2 * j]];
      const unsigned hi = (unsigned)Bm[boff + jo[2 * j + 1]];
      bw[j] = (lo | (hi << 16)) & bmask;
    }

    if (k0 + TK < K) {                  // prefetch next K tile -> global_prefetch_b8
      __builtin_prefetch(A + aoff + TK, 0, 1);
      __builtin_prefetch(Bm + boff + bstep, 0, 1);
    }
    aoff += TK;
    boff += bstep;

    __syncthreads();                    // previous iteration's fragment reads done
    *(uint4*)&As[arow * LP + asel]     = av0;
    *(uint4*)&As[arow * LP + asel + 8] = av1;
    uint4 b0; b0.x = bw[0]; b0.y = bw[1]; b0.z = bw[2]; b0.w = bw[3];
    uint4 b1; b1.x = bw[4]; b1.y = bw[5]; b1.z = bw[6]; b1.w = bw[7];
    *(uint4*)&Bs[arow * LP + asel]     = b0;
    *(uint4*)&Bs[arow * LP + asel + 8] = b1;
    __syncthreads();

    // --- fragment loads per the 16-bit 16x32 A layout:
    // lanes 0-15 hold K 0..7 (vgpr0-3) and 16..23 (vgpr4-7); lanes 16-31 hold K 8..15 / 24..31
    FragBF afr[2], bfr[4];
    const int fr = lane & 15;
    const int kb = (lane >> 4) * 8;
#pragma unroll
    for (int i = 0; i < 2; ++i) {
      const u16* p = &As[(waveM + i * 16 + fr) * LP + kb];
      afr[i].u[0] = *(const uint4*)(p);
      afr[i].u[1] = *(const uint4*)(p + 16);
    }
#pragma unroll
    for (int j = 0; j < 4; ++j) {
      const u16* p = &Bs[(waveN + j * 16 + fr) * LP + kb];
      bfr[j].u[0] = *(const uint4*)(p);
      bfr[j].u[1] = *(const uint4*)(p + 16);
    }
#pragma unroll
    for (int i = 0; i < 2; ++i)
#pragma unroll
      for (int j = 0; j < 4; ++j)
        acc[i][j] = __builtin_amdgcn_wmma_f32_16x16x32_bf16(
            false, afr[i].v, false, bfr[j].v, (short)0, acc[i][j], false, false);
  }

  // --- epilogue: C/D layout: vgpr r, lanes 0-15 -> (M=r, N=lane); lanes 16-31 -> (M=8+r)
  const int rOff = (lane >> 4) * 8;
  const int col  = lane & 15;
#pragma unroll
  for (int i = 0; i < 2; ++i) {
#pragma unroll
    for (int j = 0; j < 4; ++j) {
      const int n = n0 + waveN + j * 16 + col;
      if (n < N) {
        const float bv = (epi != 0) ? bias[n] : 0.f;
        const unsigned cbase = (unsigned)((m0 + waveM + i * 16 + rOff) * ldc + n);
#pragma unroll
        for (int r = 0; r < 8; ++r) {
          float v = acc[i][j][r] + bv;
          if (epi == 2) v = fmaxf(v, 0.f);
          C[cbase + (unsigned)(r * ldc)] = v;
        }
      }
    }
  }
}

// ---------------------------------------------------------------- row softmax (mask before /8), bf16 out
__global__ __launch_bounds__(256)
void softmax_bf16(const float* __restrict__ Sc, u16* __restrict__ P, int cols, int causal)
{
  const int row = blockIdx.x;                                   // gridDim.x == S
  const long long base = ((long long)blockIdx.y * gridDim.x + row) * cols;
  __shared__ float red[256];
  const int tid = threadIdx.x;

  float vals[4];                                                // cols == 1024
  float mx = -3.0e38f;
#pragma unroll
  for (int i = 0; i < 4; ++i) {
    const int c = tid + i * 256;
    const float s = Sc[base + c];
    const bool valid = (!causal) || (c <= row);
    const float v = valid ? s * 0.125f : -3.0e38f;              // -inf surrogate; exp underflows to 0
    vals[i] = v;
    mx = fmaxf(mx, v);
  }
  red[tid] = mx; __syncthreads();
  for (int s = 128; s > 0; s >>= 1) { if (tid < s) red[tid] = fmaxf(red[tid], red[tid + s]); __syncthreads(); }
  mx = red[0]; __syncthreads();

  float sum = 0.f;
#pragma unroll
  for (int i = 0; i < 4; ++i) { const float e = __expf(vals[i] - mx); vals[i] = e; sum += e; }
  red[tid] = sum; __syncthreads();
  for (int s = 128; s > 0; s >>= 1) { if (tid < s) red[tid] += red[tid + s]; __syncthreads(); }
  const float inv = 1.f / red[0];
#pragma unroll
  for (int i = 0; i < 4; ++i) {
    const int c = tid + i * 256;
    P[base + c] = f2bf(vals[i] * inv);
  }
}

// ---------------------------------------------------------------- residual add + LayerNorm over (S,E)
__global__ __launch_bounds__(1024)
void layernorm2d(const float* __restrict__ x, const float* __restrict__ res,
                 const float* __restrict__ w, const float* __restrict__ b,
                 float* __restrict__ y, int n)
{
  const long long base = (long long)blockIdx.x * n;
  __shared__ float s1[1024];
  __shared__ float s2[1024];
  const int tid = threadIdx.x;
  float sum = 0.f, sq = 0.f;
  for (int i = tid; i < n; i += 1024) {
    const float v = x[base + i] + res[base + i];
    sum += v; sq += v * v;
  }
  s1[tid] = sum; s2[tid] = sq; __syncthreads();
  for (int s = 512; s > 0; s >>= 1) {
    if (tid < s) { s1[tid] += s1[tid + s]; s2[tid] += s2[tid + s]; }
    __syncthreads();
  }
  const float mean = s1[0] / n;
  const float var  = s2[0] / n - mean * mean;
  const float inv  = rsqrtf(var + 1e-5f);
  for (int i = tid; i < n; i += 1024) {
    const float v = x[base + i] + res[base + i];
    y[base + i] = (v - mean) * inv * w[i] + b[i];
  }
}

// ---------------------------------------------------------------- host orchestration
extern "C" void kernel_launch(void* const* d_in, const int* in_sizes, int n_in,
                              void* d_out, int out_size, void* d_ws, size_t ws_size,
                              hipStream_t stream) {
  (void)in_sizes; (void)n_in; (void)out_size; (void)ws_size;
  const int Bz = 4, S = 1024, E = 1024, H = 16, DK = 64, FF = 2048;
  const long long BSE  = (long long)Bz * S * E;        // 4,194,304
  const long long HBSD = (long long)H * Bz * S * DK;   // 4,194,304

  const float* inputRes = (const float*)d_in[0];
  const float* outEnc   = (const float*)d_in[1];
  const float* Wq1 = (const float*)d_in[2];
  const float* Wk1 = (const float*)d_in[3];
  const float* Wv1 = (const float*)d_in[4];
  const float* Wo1 = (const float*)d_in[5];
  const float* Wq2 = (const float*)d_in[6];
  const float* Wk2 = (const float*)d_in[7];
  const float* Wv2 = (const float*)d_in[8];
  const float* Wo2 = (const float*)d_in[9];
  const float* ln1w = (const float*)d_in[10];
  const float* ln1b = (const float*)d_in[11];
  const float* ln2w = (const float*)d_in[12];
  const float* ln2b = (const float*)d_in[13];
  const float* ln3w = (const float*)d_in[14];
  const float* ln3b = (const float*)d_in[15];
  const float* W1 = (const float*)d_in[16];
  const float* b1 = (const float*)d_in[17];
  const float* W2 = (const float*)d_in[18];
  const float* b2 = (const float*)d_in[19];

  size_t off = 0;
  auto alloc = [&](size_t bytes) -> void* {
    void* p = (char*)d_ws + off;
    off += (bytes + 255) & ~(size_t)255;
    return p;
  };

  u16* xq_h   = (u16*)alloc(BSE * 2);                 // outputEncoding bf16
  u16* xkv_h  = (u16*)alloc(BSE * 2);                 // inputRes bf16
  u16* attq_h = (u16*)alloc(BSE * 2);                 // LN1 output bf16 (queries for MHA2)
  u16* Wq_h = (u16*)alloc((size_t)H * E * DK * 2);
  u16* Wk_h = (u16*)alloc((size_t)H * E * DK * 2);
  u16* Wv_h = (u16*)alloc((size_t)H * E * DK * 2);
  u16* Wo_h = (u16*)alloc((size_t)H * DK * E * 2);
  u16* W1_h = (u16*)alloc((size_t)S * FF * 2);
  u16* W2_h = (u16*)alloc((size_t)FF * S * 2);
  float* q_f = (float*)alloc(HBSD * 4);
  float* k_f = (float*)alloc(HBSD * 4);
  float* v_f = (float*)alloc(HBSD * 4);
  u16* q_h = (u16*)alloc(HBSD * 2);
  u16* k_h = (u16*)alloc(HBSD * 2);
  u16* v_h = (u16*)alloc(HBSD * 2);
  float* sc_f = (float*)alloc((size_t)Bz * S * S * 4);   // per-head scores (reused)
  u16*   pr_h = (u16*)alloc((size_t)Bz * S * S * 2);     // per-head probs  (reused)
  float* ho_f = (float*)alloc(HBSD * 4);                 // [H,B,S,DK] head outputs
  u16*   ho_h = (u16*)alloc(HBSD * 2);
  float* att1_f = (float*)alloc(BSE * 4);
  float* n1_f   = (float*)alloc(BSE * 4);
  float* att2_f = (float*)alloc(BSE * 4);
  float* n2_f   = (float*)alloc(BSE * 4);
  u16*   n2_h   = (u16*)alloc(BSE * 2);
  float* fc1_f  = (float*)alloc((size_t)Bz * E * FF * 4);
  u16*   fc1_h  = (u16*)alloc((size_t)Bz * E * FF * 2);
  float* fc2_f  = (float*)alloc(BSE * 4);

  auto cvt = [&](const float* x, u16* y, long long n) {
    cvt_f32_bf16<<<dim3(2048), dim3(256), 0, stream>>>(x, y, n);
  };
  auto gemm = [&](const u16* A, const u16* Bp, float* C, int M, int N, int K,
                  int lda, int ldbk, int ldbn, int ldc,
                  long long sA, long long sB, long long sC, int batches,
                  const float* bias, int epi) {
    dim3 g((N + TN - 1) / TN, (M + TM - 1) / TM, batches);
    wmma_gemm_bf16<<<g, dim3(256), 0, stream>>>(A, Bp, C, M, N, K,
                                                lda, ldbk, ldbn, ldc, sA, sB, sC, bias, epi);
  };

  auto mha = [&](const u16* xq, const u16* xkv, int causal, float* out_f) {
    // Q/K/V projections, batched over heads (A shared across batch via stride 0)
    gemm(xq,  Wq_h, q_f, Bz * S, DK, E, E, DK, 1, DK, 0, (long long)E * DK, (long long)Bz * S * DK, H, nullptr, 0);
    gemm(xkv, Wk_h, k_f, Bz * S, DK, E, E, DK, 1, DK, 0, (long long)E * DK, (long long)Bz * S * DK, H, nullptr, 0);
    gemm(xkv, Wv_h, v_f, Bz * S, DK, E, E, DK, 1, DK, 0, (long long)E * DK, (long long)Bz * S * DK, H, nullptr, 0);
    cvt(q_f, q_h, HBSD); cvt(k_f, k_h, HBSD); cvt(v_f, v_h, HBSD);
    for (int h = 0; h < H; ++h) {
      const u16* qh = q_h + (long long)h * Bz * S * DK;
      const u16* kh = k_h + (long long)h * Bz * S * DK;
      const u16* vh = v_h + (long long)h * Bz * S * DK;
      // scores = Q * K^T : B accessed transposed via (ldbk=1, ldbn=DK)
      gemm(qh, kh, sc_f, S, S, DK, DK, 1, DK, S,
           (long long)S * DK, (long long)S * DK, (long long)S * S, Bz, nullptr, 0);
      softmax_bf16<<<dim3(S, Bz), dim3(256), 0, stream>>>(sc_f, pr_h, S, causal);
      // out = P * V
      gemm(pr_h, vh, ho_f + (long long)h * Bz * S * DK, S, DK, S, S, DK, 1, DK,
           (long long)S * S, (long long)S * DK, (long long)S * DK, Bz, nullptr, 0);
    }
    cvt(ho_f, ho_h, HBSD);
    // head concat is a raw reshape of the contiguous [H,B,S,DK] buffer -> [B*S, E]
    gemm(ho_h, Wo_h, out_f, Bz * S, E, H * DK, H * DK, E, 1, E, 0, 0, 0, 1, nullptr, 0);
  };

  // activations + MHA1 weights -> bf16
  cvt(outEnc,   xq_h,  BSE);
  cvt(inputRes, xkv_h, BSE);
  cvt(Wq1, Wq_h, (long long)H * E * DK);
  cvt(Wk1, Wk_h, (long long)H * E * DK);
  cvt(Wv1, Wv_h, (long long)H * E * DK);
  cvt(Wo1, Wo_h, (long long)H * DK * E);
  cvt(W1, W1_h, (long long)S * FF);
  cvt(W2, W2_h, (long long)FF * S);

  // masked self-attention + residual + LN1
  mha(xq_h, xq_h, /*causal=*/1, att1_f);
  layernorm2d<<<dim3(Bz), dim3(1024), 0, stream>>>(att1_f, outEnc, ln1w, ln1b, n1_f, S * E);
  cvt(n1_f, attq_h, BSE);

  // cross-attention (K,V from inputRes) + residual to outputEncoding + LN2
  cvt(Wq2, Wq_h, (long long)H * E * DK);
  cvt(Wk2, Wk_h, (long long)H * E * DK);
  cvt(Wv2, Wv_h, (long long)H * E * DK);
  cvt(Wo2, Wo_h, (long long)H * DK * E);
  mha(attq_h, xkv_h, /*causal=*/0, att2_f);
  layernorm2d<<<dim3(Bz), dim3(1024), 0, stream>>>(att2_f, outEnc, ln2w, ln2b, n2_f, S * E);
  cvt(n2_f, n2_h, BSE);

  // FFN over sequence dim: raw reshape [B,S,E]->[B,E,S] is a free reinterpret of n2_h
  gemm(n2_h,  W1_h, fc1_f, Bz * E, FF, S,  S,  FF, 1, FF, 0, 0, 0, 1, b1, /*relu+bias*/2);
  cvt(fc1_f, fc1_h, (long long)Bz * E * FF);
  gemm(fc1_h, W2_h, fc2_f, Bz * E, S,  FF, FF, S,  1, S,  0, 0, 0, 1, b2, /*bias*/1);

  // fc reshaped back to [B,S,E] (raw) + residual att2_norm + LN3 -> output
  layernorm2d<<<dim3(Bz), dim3(1024), 0, stream>>>(fc2_f, n2_f, ln3w, ln3b, (float*)d_out, S * E);
}
